// GateL0RDCell_84018150245162
// MI455X (gfx1250) — matrix operand order repaired and checked by reading, not compiled, not run on hardware
//
#include <hip/hip_runtime.h>

typedef __attribute__((ext_vector_type(8)))  __bf16 v8bf;
typedef __attribute__((ext_vector_type(16))) __bf16 v16bf;
typedef __attribute__((ext_vector_type(8)))  float  v8f;
typedef __attribute__((ext_vector_type(4)))  unsigned int u32x4;
typedef __attribute__((ext_vector_type(4)))  int i32x4;
typedef __attribute__((ext_vector_type(8)))  int i32x8;

#define B_ROWS  16384
#define IN_DIM  512
#define LAT     1024
#define OUT_DIM 512
#define GDIM    1536   // IN_DIM + LAT

#define KTILE 32
#define NKT   (GDIM / KTILE)   // 48
#define MBLK  128
#define NBLK  64
#define LDS_STRIDE 40          // halves per row: 32 data + 4-DWORD TDM pad (80B rows)

__device__ __forceinline__ unsigned short f32_to_bf16(float f) {
  unsigned u = __float_as_uint(f);
  u += 0x7FFFu + ((u >> 16) & 1u);       // round to nearest even
  return (unsigned short)(u >> 16);
}

__device__ __forceinline__ v16bf cat8(v8bf lo, v8bf hi) {
  return __builtin_shufflevector(lo, hi, 0,1,2,3,4,5,6,7,8,9,10,11,12,13,14,15);
}

__device__ __forceinline__ float sigmoidf(float x) {
  return 1.0f / (1.0f + __expf(-x));
}

// ---- Tensor Data Mover: load a 2D bf16 tile [MBLK rows x KTILE halves] into LDS,
//      padding each 64B row to 80B (pad_interval=16 DWORDs, pad_amount=4 DWORDs).
//      lds_dst is a generic pointer to the LDS destination; its low 32 bits are the
//      LDS byte offset (gfx1250 LDS aperture layout), and the ptrtoint marks the
//      LDS array as escaped so the compiler keeps the consumer-side ds_loads.
__device__ __forceinline__ void tdm_load_tile(const unsigned short* lds_dst,
                                              const unsigned short* gsrc, int lda) {
  unsigned lds_off = (unsigned)(size_t)lds_dst;          // LDS byte address
  unsigned long long ga = (unsigned long long)(size_t)gsrc;
  u32x4 g0;
  g0[0] = 1u;                                            // count=1 (valid user D#)
  g0[1] = lds_off;                                       // lds_addr (bytes)
  g0[2] = (unsigned)ga;                                  // global_addr[31:0]
  g0[3] = (unsigned)((ga >> 32) & 0x01FFFFFFu) | (2u << 30); // global_addr[56:32] | type=2
  i32x8 g1;
  g1[0] = (int)((1u << 16)        // data_size = 1 -> 2 bytes
              | (1u << 20)        // pad_enable
              | (3u << 22)        // pad_interval: 16 DWORDs (one 64B row)
              | (3u << 25));      // pad_amount: 4 DWORDs (16B)
  g1[1] = (int)(((unsigned)lda & 0xFFFFu) << 16);        // tensor_dim0[15:0]
  g1[2] = (int)(((unsigned)B_ROWS & 0xFFFFu) << 16);     // tensor_dim0[31:16]=0 | tensor_dim1[15:0]
  g1[3] = (int)(((unsigned)KTILE) << 16);                // tensor_dim1[31:16]=0 | tile_dim0=32
  g1[4] = MBLK;                                          // tile_dim1=128 | tile_dim2=0
  g1[5] = lda;                                           // tensor_dim0_stride[31:0]
  g1[6] = 0;                                             // stride0[47:32] | stride1[15:0]
  g1[7] = 0;
  i32x4 gz = {0, 0, 0, 0};
#if __has_include(<hip/amd_detail/amd_gfx1250_TDM.h>)
  i32x8 gz8 = {0, 0, 0, 0, 0, 0, 0, 0};
  __builtin_amdgcn_tensor_load_to_lds(g0, g1, gz, gz, gz8, 0);
#else
  __builtin_amdgcn_tensor_load_to_lds(g0, g1, gz, gz, 0);
#endif
}

// fp32 [K x N] row-major -> bf16 [N x K] row-major (B-fragments contiguous in K)
__global__ void wt_bf16_kernel(const float* __restrict__ W,
                               unsigned short* __restrict__ WT,
                               int K, int N) {
  int idx = blockIdx.x * blockDim.x + threadIdx.x;
  if (idx >= K * N) return;
  int k = idx / N, n = idx % N;
  WT[(size_t)n * K + k] = f32_to_bf16(W[idx]);
}

// fp32 -> bf16 elementwise (activations), 4 elems/thread
__global__ void act_bf16_kernel(const float* __restrict__ src,
                                unsigned short* __restrict__ dst, int n4) {
  int i = blockIdx.x * blockDim.x + threadIdx.x;
  if (i >= n4) return;
  const float4 v = reinterpret_cast<const float4*>(src)[i];
  uint2 o;
  o.x = (unsigned)f32_to_bf16(v.x) | ((unsigned)f32_to_bf16(v.y) << 16);
  o.y = (unsigned)f32_to_bf16(v.z) | ((unsigned)f32_to_bf16(v.w) << 16);
  reinterpret_cast<uint2*>(dst)[i] = o;
}

// ---------------- Phase 1: i = gr@Wg+bg ; r = gr@Wr+br ; gate epilogue ----------------
__global__ __launch_bounds__(256)
void gatel0rd_phase1(const unsigned short* __restrict__ xb,   // [B, IN_DIM] bf16
                     const unsigned short* __restrict__ hb,   // [B, LAT] bf16
                     const float* __restrict__ hprev,
                     const float* __restrict__ noise,
                     const unsigned short* __restrict__ wgT,
                     const unsigned short* __restrict__ wrT,
                     const float* __restrict__ bg,
                     const float* __restrict__ br,
                     float* __restrict__ h_out,
                     float* __restrict__ theta_out,
                     unsigned short* __restrict__ htb)        // [B, LAT] bf16
{
  __shared__ unsigned short sA[2][MBLK * LDS_STRIDE];

  const int t    = threadIdx.x;
  const int lane = t & 31;
  const int wave = t >> 5;
  const int wm   = wave & 3;   // M sub-block of 32
  const int wn   = wave >> 2;  // N sub-block of 32
  const int lo16 = lane & 15;
  const int hi   = lane >> 4;

  const int mBase = blockIdx.x * MBLK;
  const int nBase = blockIdx.y * NBLK;

  const v8f zero = {0.f,0.f,0.f,0.f,0.f,0.f,0.f,0.f};
  v8f accG[2][2], accR[2][2];
  #pragma unroll
  for (int i = 0; i < 2; ++i)
    #pragma unroll
    for (int j = 0; j < 2; ++j) { accG[i][j] = zero; accR[i][j] = zero; }

  const bool issuer = (wave == 0);
  if (issuer) tdm_load_tile(&sA[0][0], xb + (size_t)mBase * IN_DIM, IN_DIM);

  for (int kt = 0; kt < NKT; ++kt) {
    const int cur = kt & 1;
    const int k0  = kt * KTILE;
    if (issuer) {
      if (kt + 1 < NKT) {
        const int kk = k0 + KTILE;
        const unsigned short* src; int lda;
        if (kk < IN_DIM) { src = xb + (size_t)mBase * IN_DIM + kk;            lda = IN_DIM; }
        else             { src = hb + (size_t)mBase * LAT    + (kk - IN_DIM); lda = LAT; }
        tdm_load_tile(&sA[cur ^ 1][0], src, lda);
        __builtin_amdgcn_s_wait_tensorcnt((short)1);   // tile kt complete (in-order)
      } else {
        __builtin_amdgcn_s_wait_tensorcnt((short)0);
      }
    }
    __syncthreads();                      // tile kt visible to all waves
    asm volatile("" ::: "memory");        // LDS was written by the TDM: force reloads

    v16bf fg[2], fr[2];
    #pragma unroll
    for (int nn = 0; nn < 2; ++nn) {
      int n = nBase + wn * 32 + nn * 16 + lo16;
      const unsigned short* pg = wgT + (size_t)n * GDIM + k0 + hi * 16;
      const unsigned short* pr = wrT + (size_t)n * GDIM + k0 + hi * 16;
      fg[nn] = cat8(*(const v8bf*)pg, *(const v8bf*)(pg + 8));
      fr[nn] = cat8(*(const v8bf*)pr, *(const v8bf*)(pr + 8));
      if (kt + 1 < NKT) {
        __builtin_prefetch(pg + KTILE, 0, 3);
        __builtin_prefetch(pr + KTILE, 0, 3);
      }
    }

    #pragma unroll
    for (int mm = 0; mm < 2; ++mm) {
      int row = wm * 32 + mm * 16 + lo16;
      const unsigned short* pa = &sA[cur][row * LDS_STRIDE + hi * 16];
      v16bf a = cat8(*(const v8bf*)pa, *(const v8bf*)(pa + 8));
      #pragma unroll
      for (int nn = 0; nn < 2; ++nn) {
        accG[mm][nn] = __builtin_amdgcn_wmma_f32_16x16x32_bf16(
            false, a, false, fg[nn], (short)0, accG[mm][nn], false, false);
        accR[mm][nn] = __builtin_amdgcn_wmma_f32_16x16x32_bf16(
            false, a, false, fr[nn], (short)0, accR[mm][nn], false, false);
      }
    }
    __syncthreads();   // done reading sA[cur]; TDM may refill it next iteration
  }

  // Lambda = max(tanh(i - noise), 0); h = Lam*tanh(r) + (1-Lam)*h_prev; Theta = Lam>0
  #pragma unroll
  for (int mm = 0; mm < 2; ++mm) {
    #pragma unroll
    for (int nn = 0; nn < 2; ++nn) {
      int n = nBase + wn * 32 + nn * 16 + lo16;
      float bgv = bg[n], brv = br[n];
      #pragma unroll
      for (int j = 0; j < 8; ++j) {
        int m = mBase + wm * 32 + mm * 16 + hi * 8 + j;
        size_t idx = (size_t)m * LAT + n;
        float i_t = accG[mm][nn][j] + bgv;
        float lam = fmaxf(tanhf(i_t - noise[idx]), 0.f);
        float hh  = tanhf(accR[mm][nn][j] + brv);
        float ht  = lam * hh + (1.f - lam) * hprev[idx];
        h_out[idx]     = ht;
        theta_out[idx] = (lam > 0.f) ? 1.f : 0.f;
        htb[idx]       = f32_to_bf16(ht);
      }
    }
  }
}

// ---------------- Phase 2: y = tanh(xh@Wy+by) * sigmoid(xh@Wo+bo) ----------------
__global__ __launch_bounds__(256)
void gatel0rd_phase2(const unsigned short* __restrict__ xb,
                     const unsigned short* __restrict__ htb,
                     const unsigned short* __restrict__ wyT,
                     const unsigned short* __restrict__ woT,
                     const float* __restrict__ by,
                     const float* __restrict__ bo,
                     float* __restrict__ y_out)
{
  __shared__ unsigned short sA[2][MBLK * LDS_STRIDE];

  const int t    = threadIdx.x;
  const int lane = t & 31;
  const int wave = t >> 5;
  const int wm   = wave & 3;
  const int wn   = wave >> 2;
  const int lo16 = lane & 15;
  const int hi   = lane >> 4;

  const int mBase = blockIdx.x * MBLK;
  const int nBase = blockIdx.y * NBLK;

  const v8f zero = {0.f,0.f,0.f,0.f,0.f,0.f,0.f,0.f};
  v8f accY[2][2], accO[2][2];
  #pragma unroll
  for (int i = 0; i < 2; ++i)
    #pragma unroll
    for (int j = 0; j < 2; ++j) { accY[i][j] = zero; accO[i][j] = zero; }

  const bool issuer = (wave == 0);
  if (issuer) tdm_load_tile(&sA[0][0], xb + (size_t)mBase * IN_DIM, IN_DIM);

  for (int kt = 0; kt < NKT; ++kt) {
    const int cur = kt & 1;
    const int k0  = kt * KTILE;
    if (issuer) {
      if (kt + 1 < NKT) {
        const int kk = k0 + KTILE;
        const unsigned short* src; int lda;
        if (kk < IN_DIM) { src = xb  + (size_t)mBase * IN_DIM + kk;            lda = IN_DIM; }
        else             { src = htb + (size_t)mBase * LAT    + (kk - IN_DIM); lda = LAT; }
        tdm_load_tile(&sA[cur ^ 1][0], src, lda);
        __builtin_amdgcn_s_wait_tensorcnt((short)1);
      } else {
        __builtin_amdgcn_s_wait_tensorcnt((short)0);
      }
    }
    __syncthreads();
    asm volatile("" ::: "memory");

    v16bf fy[2], fo[2];
    #pragma unroll
    for (int nn = 0; nn < 2; ++nn) {
      int n = nBase + wn * 32 + nn * 16 + lo16;
      const unsigned short* py = wyT + (size_t)n * GDIM + k0 + hi * 16;
      const unsigned short* po = woT + (size_t)n * GDIM + k0 + hi * 16;
      fy[nn] = cat8(*(const v8bf*)py, *(const v8bf*)(py + 8));
      fo[nn] = cat8(*(const v8bf*)po, *(const v8bf*)(po + 8));
      if (kt + 1 < NKT) {
        __builtin_prefetch(py + KTILE, 0, 3);
        __builtin_prefetch(po + KTILE, 0, 3);
      }
    }

    #pragma unroll
    for (int mm = 0; mm < 2; ++mm) {
      int row = wm * 32 + mm * 16 + lo16;
      const unsigned short* pa = &sA[cur][row * LDS_STRIDE + hi * 16];
      v16bf a = cat8(*(const v8bf*)pa, *(const v8bf*)(pa + 8));
      #pragma unroll
      for (int nn = 0; nn < 2; ++nn) {
        accY[mm][nn] = __builtin_amdgcn_wmma_f32_16x16x32_bf16(
            false, a, false, fy[nn], (short)0, accY[mm][nn], false, false);
        accO[mm][nn] = __builtin_amdgcn_wmma_f32_16x16x32_bf16(
            false, a, false, fo[nn], (short)0, accO[mm][nn], false, false);
      }
    }
    __syncthreads();
  }

  #pragma unroll
  for (int mm = 0; mm < 2; ++mm) {
    #pragma unroll
    for (int nn = 0; nn < 2; ++nn) {
      int n = nBase + wn * 32 + nn * 16 + lo16;
      float byv = by[n], bov = bo[n];
      #pragma unroll
      for (int j = 0; j < 8; ++j) {
        int m = mBase + wm * 32 + mm * 16 + hi * 8 + j;
        size_t idx = (size_t)m * OUT_DIM + n;
        float yh = tanhf(accY[mm][nn][j] + byv);
        float og = sigmoidf(accO[mm][nn][j] + bov);
        y_out[idx] = yh * og;
      }
    }
  }
}

extern "C" void kernel_launch(void* const* d_in, const int* in_sizes, int n_in,
                              void* d_out, int out_size, void* d_ws, size_t ws_size,
                              hipStream_t stream) {
  const float* x     = (const float*)d_in[0];
  const float* hprev = (const float*)d_in[1];
  const float* noise = (const float*)d_in[2];
  const float* Wg    = (const float*)d_in[3];
  const float* bg    = (const float*)d_in[4];
  const float* Wr    = (const float*)d_in[5];
  const float* br    = (const float*)d_in[6];
  const float* Wy    = (const float*)d_in[7];
  const float* by    = (const float*)d_in[8];
  const float* Wo    = (const float*)d_in[9];
  const float* bo    = (const float*)d_in[10];

  float* out    = (float*)d_out;
  float* y_out  = out;                                  // [B, OUT_DIM]
  float* h_out  = y_out + (size_t)B_ROWS * OUT_DIM;     // [B, LAT]
  float* th_out = h_out + (size_t)B_ROWS * LAT;         // [B, LAT]

  unsigned short* wgT = (unsigned short*)d_ws;          // [LAT, GDIM] bf16
  unsigned short* wrT = wgT + (size_t)LAT * GDIM;
  unsigned short* wyT = wrT + (size_t)LAT * GDIM;       // [OUT_DIM, GDIM]
  unsigned short* woT = wyT + (size_t)OUT_DIM * GDIM;
  unsigned short* xb  = woT + (size_t)OUT_DIM * GDIM;   // [B, IN_DIM] bf16
  unsigned short* hb  = xb  + (size_t)B_ROWS * IN_DIM;  // [B, LAT] bf16
  unsigned short* htb = hb  + (size_t)B_ROWS * LAT;     // [B, LAT] bf16

  const int tot1 = GDIM * LAT;
  const int tot2 = GDIM * OUT_DIM;
  wt_bf16_kernel<<<(tot1 + 255) / 256, 256, 0, stream>>>(Wg, wgT, GDIM, LAT);
  wt_bf16_kernel<<<(tot1 + 255) / 256, 256, 0, stream>>>(Wr, wrT, GDIM, LAT);
  wt_bf16_kernel<<<(tot2 + 255) / 256, 256, 0, stream>>>(Wy, wyT, GDIM, OUT_DIM);
  wt_bf16_kernel<<<(tot2 + 255) / 256, 256, 0, stream>>>(Wo, woT, GDIM, OUT_DIM);

  const int nx4 = B_ROWS * IN_DIM / 4;
  const int nh4 = B_ROWS * LAT / 4;
  act_bf16_kernel<<<(nx4 + 255) / 256, 256, 0, stream>>>(x, xb, nx4);
  act_bf16_kernel<<<(nh4 + 255) / 256, 256, 0, stream>>>(hprev, hb, nh4);

  dim3 g1(B_ROWS / MBLK, LAT / NBLK);      // 128 x 16
  gatel0rd_phase1<<<g1, 256, 0, stream>>>(xb, hb, hprev, noise, wgT, wrT, bg, br,
                                          h_out, th_out, htb);

  dim3 g2(B_ROWS / MBLK, OUT_DIM / NBLK);  // 128 x 8
  gatel0rd_phase2<<<g2, 256, 0, stream>>>(xb, htb, wyT, woT, by, bo, y_out);
}
